// SE3DipolePredictor_47313359733285
// MI455X (gfx1250) — matrix operand order repaired and checked by reading, not compiled, not run on hardware
//
#include <hip/hip_runtime.h>
#include <hip/hip_bf16.h>

typedef _Float16 v16h __attribute__((ext_vector_type(16)));
typedef _Float16 v8h  __attribute__((ext_vector_type(8)));
typedef float    v8f  __attribute__((ext_vector_type(8)));

// fast swish: x * sigmoid(x) using v_exp_f32 + v_rcp_f32 (no IEEE div expansion)
__device__ __forceinline__ float fast_swish(float v) {
    float e = __expf(-v);
    return v * __builtin_amdgcn_rcpf(1.0f + e);
}

// ---------------------------------------------------------------------------
// utility
// ---------------------------------------------------------------------------
__global__ void zero_f32_kernel(float* __restrict__ p, long long n) {
    long long i = (long long)blockIdx.x * blockDim.x + threadIdx.x;
    if (i < n) p[i] = 0.0f;
}

// ---------------------------------------------------------------------------
// Kernel 1: edge geometry -> cosine basis -> layer1 swish -> f16 rows [*,128]
// rows for padded (nonexistent) edges are written as zeros so the WMMA GEMM
// can run on 128-row tiles unconditionally.
// ---------------------------------------------------------------------------
__global__ void edge_front_kernel(const float* __restrict__ pos,
                                  const int*   __restrict__ ei,
                                  const float* __restrict__ rw1,
                                  const float* __restrict__ rb1,
                                  _Float16*    __restrict__ h1,
                                  long long chunkStart, int chunkEdges,
                                  int paddedChunk, long long E) {
    __shared__ float srw1[300];
    __shared__ float srb1[100];
    for (int i = threadIdx.x; i < 300; i += blockDim.x) srw1[i] = rw1[i];
    for (int i = threadIdx.x; i < 100; i += blockDim.x) srb1[i] = rb1[i];
    __syncthreads();

    int li = blockIdx.x * blockDim.x + threadIdx.x;
    if (li >= paddedChunk) return;
    _Float16* row = h1 + (size_t)li * 128;

    if (li >= chunkEdges) {
        for (int j = 0; j < 128; ++j) row[j] = (_Float16)0.0f;
        return;
    }
    long long e = chunkStart + li;
    int s = ei[e];
    int d = ei[E + e];
    float dx = pos[3 * s + 0] - pos[3 * d + 0];
    float dy = pos[3 * s + 1] - pos[3 * d + 1];
    float dz = pos[3 * s + 2] - pos[3 * d + 2];
    float dist = sqrtf(dx * dx + dy * dy + dz * dz + 1e-12f);

    // cosine basis: centers {0, 1.5, 3.0}, step 1.5
    float b[3];
#pragma unroll
    for (int k = 0; k < 3; ++k) {
        float diff = (dist - 1.5f * (float)k) * (1.0f / 1.5f);
        float c = __cosf(1.57079632679f * diff);
        b[k] = (fabsf(diff) < 1.0f) ? c * c : 0.0f;
    }
    for (int j = 0; j < 100; ++j) {
        float v = b[0] * srw1[j] + b[1] * srw1[100 + j] + b[2] * srw1[200 + j] + srb1[j];
        row[j] = (_Float16)fast_swish(v);
    }
    for (int j = 100; j < 128; ++j) row[j] = (_Float16)0.0f;
}

// ---------------------------------------------------------------------------
// Kernel 2: layer2 GEMM via WMMA (f16 in, f32 acc) fused with bias+swish and
// layer3 contraction (100 -> 2 path weights).
// Block: 256 threads = 8 waves; each wave computes 16 edge-rows.
// K = 128 (4 x 32), N = 112 (7 x 16) with zero padding beyond 100.
// ---------------------------------------------------------------------------
__launch_bounds__(256)
__global__ void gemm_w_kernel(const _Float16* __restrict__ h1,
                              const float* __restrict__ rw2,   // [100,100] row-major (k,n)
                              const float* __restrict__ rb2,   // [100]
                              const float* __restrict__ rw3,   // [100,2]
                              float* __restrict__ wbuf) {      // [*,2]
    __shared__ _Float16 Bs[112 * 128];   // transposed: Bs[n*128 + k]
    __shared__ float    rb2s[112];
    __shared__ float    rw3s[112 * 2];

    int tid = threadIdx.x;
    for (int idx = tid; idx < 112 * 128; idx += 256) {
        int n = idx >> 7;
        int k = idx & 127;
        float v = (n < 100 && k < 100) ? rw2[k * 100 + n] : 0.0f;
        Bs[idx] = (_Float16)v;
    }
    for (int n = tid; n < 112; n += 256) {
        rb2s[n]        = (n < 100) ? rb2[n]         : 0.0f;
        rw3s[2 * n]    = (n < 100) ? rw3[2 * n]     : 0.0f;
        rw3s[2 * n + 1]= (n < 100) ? rw3[2 * n + 1] : 0.0f;
    }
    __syncthreads();

    int wave = tid >> 5;
    int lane = tid & 31;
    int m    = lane & 15;        // row (A/M) or column (B/N, C/N) within tile
    int hi   = lane >> 4;        // lane-half selector
    int rowBase = blockIdx.x * 128 + wave * 16;

    // --- A fragments: ISA 16-bit A 16x32 layout ---
    // lanes 0-15 : halves 0..7 = K 0..7 , halves 8..15 = K 16..23
    // lanes 16-31: halves 0..7 = K 8..15, halves 8..15 = K 24..31
    const _Float16* arow = h1 + (size_t)(rowBase + m) * 128;
    v16h afrag[4];
#pragma unroll
    for (int kb = 0; kb < 4; ++kb) {
        int ko = kb * 32 + hi * 8;
        v8h lo  = *(const v8h*)(arow + ko);
        v8h hi8 = *(const v8h*)(arow + ko + 16);
        afrag[kb] = __builtin_shufflevector(lo, hi8, 0, 1, 2, 3, 4, 5, 6, 7,
                                            8, 9, 10, 11, 12, 13, 14, 15);
    }

    float p0[8], p1[8];
#pragma unroll
    for (int r = 0; r < 8; ++r) { p0[r] = 0.0f; p1[r] = 0.0f; }

    for (int nt = 0; nt < 7; ++nt) {
        int n = nt * 16 + m;
        const _Float16* bp = &Bs[n * 128];
        v8f c = {0.f, 0.f, 0.f, 0.f, 0.f, 0.f, 0.f, 0.f};
#pragma unroll
        for (int kb = 0; kb < 4; ++kb) {
            // B 32x16 layout: lanes 0-15 hold K 0..7,16..23, lanes 16-31 hold K 8..15,24..31
            int kh = kb * 32 + hi * 8;
            v8h blo = *(const v8h*)(bp + kh);
            v8h bhi = *(const v8h*)(bp + kh + 16);
            v16h bfrag = __builtin_shufflevector(blo, bhi, 0, 1, 2, 3, 4, 5, 6, 7,
                                                 8, 9, 10, 11, 12, 13, 14, 15);
            c = __builtin_amdgcn_wmma_f32_16x16x32_f16(
                    false, afrag[kb], false, bfrag, (short)0, c, false, false);
        }
        // epilogue: bias + swish, contract with rw3 (layer 3)
        float bn = rb2s[n];
        float r0 = rw3s[2 * n];
        float r1 = rw3s[2 * n + 1];
#pragma unroll
        for (int r = 0; r < 8; ++r) {
            float sw = fast_swish(c[r] + bn);
            p0[r] += sw * r0;
            p1[r] += sw * r1;
        }
    }

    // Reduce over the 16 lanes of each half (each lane held disjoint N columns)
#pragma unroll
    for (int r = 0; r < 8; ++r) {
#pragma unroll
        for (int off = 1; off < 16; off <<= 1) {
            p0[r] += __shfl_xor(p0[r], off, 32);
            p1[r] += __shfl_xor(p1[r], off, 32);
        }
    }
    // C layout: lanes 0-15 hold M 0..7, lanes 16-31 hold M 8..15
    if (m == 0) {
        int base = rowBase + hi * 8;
#pragma unroll
        for (int r = 0; r < 8; ++r) {
            wbuf[2 * (base + r) + 0] = p0[r];
            wbuf[2 * (base + r) + 1] = p1[r];
        }
    }
}

// ---------------------------------------------------------------------------
// Kernel 3: tensor-product message + scatter-add to destination nodes
// ---------------------------------------------------------------------------
__global__ void message_scatter_kernel(const float* __restrict__ pos,
                                       const float* __restrict__ z,
                                       const float* __restrict__ ea,
                                       const int*   __restrict__ ei,
                                       const float* __restrict__ wbuf,
                                       float* __restrict__ agg,
                                       long long chunkStart, int chunkEdges,
                                       long long E) {
    int li = blockIdx.x * blockDim.x + threadIdx.x;
    if (li >= chunkEdges) return;
    long long e = chunkStart + li;
    int s = ei[e];
    int d = ei[E + e];
    float a  = ea[e];
    float f0 = wbuf[2 * li + 0] * a;
    float f1 = wbuf[2 * li + 1] * a;
    atomicAdd(&agg[4 * d + 0], f0 * pos[3 * s + 0]);
    atomicAdd(&agg[4 * d + 1], f0 * pos[3 * s + 1]);
    atomicAdd(&agg[4 * d + 2], f0 * pos[3 * s + 2]);
    atomicAdd(&agg[4 * d + 3], f1 * z[s]);
}

// ---------------------------------------------------------------------------
// Kernel 4: per-node MLP (4->32->16) + atomic pooled sums/counts per graph
// ---------------------------------------------------------------------------
__global__ void node_mlp_kernel(const float* __restrict__ agg,
                                const int*   __restrict__ batch,
                                const float* __restrict__ lw1,
                                const float* __restrict__ lb1,
                                const float* __restrict__ lw2,
                                const float* __restrict__ lb2,
                                float* __restrict__ sums,
                                float* __restrict__ cnt,
                                float invnorm, int N) {
    __shared__ float slw1[128], slb1[32], slw2[512], slb2[16];
    int t = threadIdx.x;
    for (int i = t; i < 128; i += blockDim.x) slw1[i] = lw1[i];
    for (int i = t; i < 32;  i += blockDim.x) slb1[i] = lb1[i];
    for (int i = t; i < 512; i += blockDim.x) slw2[i] = lw2[i];
    for (int i = t; i < 16;  i += blockDim.x) slb2[i] = lb2[i];
    __syncthreads();

    int n = blockIdx.x * blockDim.x + t;
    if (n >= N) return;
    float a[4];
#pragma unroll
    for (int c = 0; c < 4; ++c) {
        float v = agg[4 * n + c] * invnorm;
        a[c] = v > 0.0f ? v : 0.0f;
    }
    float h32[32];
#pragma unroll
    for (int j = 0; j < 32; ++j) {
        float v = slb1[j];
#pragma unroll
        for (int c = 0; c < 4; ++c) v += a[c] * slw1[c * 32 + j];
        h32[j] = v > 0.0f ? v : 0.0f;
    }
    int b = batch[n];
#pragma unroll
    for (int k = 0; k < 16; ++k) {
        float v = slb2[k];
#pragma unroll
        for (int j = 0; j < 32; ++j) v += h32[j] * slw2[j * 16 + k];
        v = v > 0.0f ? v : 0.0f;
        atomicAdd(&sums[16 * b + k], v);
    }
    atomicAdd(&cnt[b], 1.0f);
}

// ---------------------------------------------------------------------------
// Kernel 5: mean pool + head MLP (16->32->3)
// ---------------------------------------------------------------------------
__global__ void head_kernel(const float* __restrict__ sums,
                            const float* __restrict__ cnt,
                            const float* __restrict__ fw1,
                            const float* __restrict__ fb1,
                            const float* __restrict__ fw2,
                            const float* __restrict__ fb2,
                            float* __restrict__ out, int G) {
    int g = blockIdx.x * blockDim.x + threadIdx.x;
    if (g >= G) return;
    float c = cnt[g];
    c = c > 1.0f ? c : 1.0f;
    float p[16];
#pragma unroll
    for (int k = 0; k < 16; ++k) p[k] = sums[16 * g + k] / c;
    float t[32];
#pragma unroll
    for (int j = 0; j < 32; ++j) {
        float v = fb1[j];
#pragma unroll
        for (int k = 0; k < 16; ++k) v += p[k] * fw1[k * 32 + j];
        t[j] = v > 0.0f ? v : 0.0f;
    }
#pragma unroll
    for (int o = 0; o < 3; ++o) {
        float v = fb2[o];
#pragma unroll
        for (int j = 0; j < 32; ++j) v += t[j] * fw2[j * 3 + o];
        out[3 * g + o] = v;
    }
}

// ---------------------------------------------------------------------------
// host launcher
// ---------------------------------------------------------------------------
static inline size_t align256(size_t x) { return (x + 255) & ~(size_t)255; }

extern "C" void kernel_launch(void* const* d_in, const int* in_sizes, int n_in,
                              void* d_out, int out_size, void* d_ws, size_t ws_size,
                              hipStream_t stream) {
    const float* pos = (const float*)d_in[0];
    const float* z   = (const float*)d_in[1];
    const float* ea  = (const float*)d_in[2];
    const int*   ei  = (const int*)d_in[3];
    const int*   bat = (const int*)d_in[4];
    const float* rw1 = (const float*)d_in[5];
    const float* rb1 = (const float*)d_in[6];
    const float* rw2 = (const float*)d_in[7];
    const float* rb2 = (const float*)d_in[8];
    const float* rw3 = (const float*)d_in[9];
    const float* lw1 = (const float*)d_in[10];
    const float* lb1 = (const float*)d_in[11];
    const float* lw2 = (const float*)d_in[12];
    const float* lb2 = (const float*)d_in[13];
    const float* fw1 = (const float*)d_in[14];
    const float* fb1 = (const float*)d_in[15];
    const float* fw2 = (const float*)d_in[16];
    const float* fb2 = (const float*)d_in[17];
    (void)n_in;

    const long long N = in_sizes[1];      // z has N elements
    const long long E = in_sizes[2];      // edge_attr has E elements
    const int G = out_size / 3;
    float* out = (float*)d_out;

    // workspace layout
    char* wsb = (char*)d_ws;
    size_t off = 0;
    float* agg  = (float*)(wsb + off); off = align256(off + (size_t)N * 4 * sizeof(float));
    float* sums = (float*)(wsb + off);
    float* cnt  = sums + (size_t)G * 16;
    off = align256(off + (size_t)G * 17 * sizeof(float));
    size_t avail = (ws_size > off) ? (ws_size - off) : 0;

    // per padded edge: 128 f16 (256 B) for h1 + 2 f32 (8 B) for w
    long long CEp = (long long)(avail / 264) & ~127LL;
    long long Epad = (E + 127) & ~127LL;
    if (CEp > Epad) CEp = Epad;
    if (CEp > 131072) CEp = 131072;
    if (CEp < 128) CEp = 128;

    float*    wbuf = (float*)(wsb + off);
    off = align256(off + (size_t)CEp * 2 * sizeof(float));
    _Float16* h1   = (_Float16*)(wsb + off);

    // zero accumulators
    {
        long long n0 = N * 4;
        zero_f32_kernel<<<(unsigned)((n0 + 255) / 256), 256, 0, stream>>>(agg, n0);
        long long n1 = (long long)G * 17;
        zero_f32_kernel<<<(unsigned)((n1 + 255) / 256), 256, 0, stream>>>(sums, n1);
    }

    // chunked edge pipeline
    for (long long cs = 0; cs < E; cs += CEp) {
        long long rem = E - cs;
        int chunkEdges = (int)(rem < CEp ? rem : CEp);
        int padded = (chunkEdges + 127) & ~127;

        edge_front_kernel<<<(padded + 255) / 256, 256, 0, stream>>>(
            pos, ei, rw1, rb1, h1, cs, chunkEdges, padded, E);

        gemm_w_kernel<<<padded / 128, 256, 0, stream>>>(
            h1, rw2, rb2, rw3, wbuf);

        message_scatter_kernel<<<(chunkEdges + 255) / 256, 256, 0, stream>>>(
            pos, z, ea, ei, wbuf, agg, cs, chunkEdges, E);
    }

    float invnorm = sqrtf((float)N / (float)E);
    node_mlp_kernel<<<(unsigned)((N + 255) / 256), 256, 0, stream>>>(
        agg, bat, lw1, lb1, lw2, lb2, sums, cnt, invnorm, (int)N);

    head_kernel<<<(G + 63) / 64, 64, 0, stream>>>(
        sums, cnt, fw1, fb1, fw2, fb2, out, G);
}
